// MultiHeadAttention_8899172237560
// MI455X (gfx1250) — matrix operand order repaired and checked by reading, not compiled, not run on hardware
//
#include <hip/hip_runtime.h>

typedef _Float16 h8  __attribute__((ext_vector_type(8)));
typedef _Float16 h16 __attribute__((ext_vector_type(16)));
typedef float    f8  __attribute__((ext_vector_type(8)));
typedef unsigned int u32x4 __attribute__((ext_vector_type(4)));
typedef int i32x4 __attribute__((ext_vector_type(4)));
typedef int i32x8 __attribute__((ext_vector_type(8)));

#define LDS_AS __attribute__((address_space(3)))

#define B_  2
#define T_  4096
#define D_  512
#define H_  8
#define DH_ 64
#define M_  (B_ * T_)                      // 8192 rows in flattened GEMMs
#define NX_ ((size_t)M_ * D_)              // 4,194,304 elems per activation
#define NW_ ((size_t)D_ * D_)              // 262,144 elems per weight

// ---------------------------------------------------------------------------
// WMMA helper: D = A(16x32 f16) * B(32x16 f16) + C(16x16 f32)
// ---------------------------------------------------------------------------
static __device__ __forceinline__ f8 wmma_f16(h16 a, h16 b, f8 c) {
  return __builtin_amdgcn_wmma_f32_16x16x32_f16(false, a, false, b,
                                                (short)0, c, false, false);
}

// A-fragment (16x32 f16) from global: per-lane elements base[0..15]
static __device__ __forceinline__ h16 load_a_f16(const _Float16* base) {
  h8 lo = *(const h8*)base;
  h8 hi = *(const h8*)(base + 8);
  h16 r;
#pragma unroll
  for (int e = 0; e < 8; ++e) { r[e] = lo[e]; r[e + 8] = hi[e]; }
  return r;
}
// Same A-fragment but from LDS via an explicit addrspace(3) pointer so the
// backend is guaranteed to emit ds_load_b128 (byte_off = LDS byte offset).
static __device__ __forceinline__ h16 load_a_lds(unsigned byte_off) {
  const LDS_AS h8* p = (const LDS_AS h8*)(size_t)byte_off;
  h8 lo = p[0];
  h8 hi = p[1];          // +8 halves
  h16 r;
#pragma unroll
  for (int e = 0; e < 8; ++e) { r[e] = lo[e]; r[e + 8] = hi[e]; }
  return r;
}
// B-fragment (32x16 f16): 16 contiguous K values per lane
static __device__ __forceinline__ h16 load_b_f16(const _Float16* base) {
  return *(const h16*)base;  // 32-byte aligned
}

static __device__ __forceinline__ unsigned rfl(unsigned x) {
  return (unsigned)__builtin_amdgcn_readfirstlane((int)x);
}

// ---------------------------------------------------------------------------
// Tensor Data Mover: 2D tile (f16 elements) global -> LDS.
// D# per CDNA5 ISA: group0 {count=1, lds_addr, global_addr,
// type=2}; group1 {data_size=1(2B), tensor_dim0/1, tile_dim0/1, dim0 stride}.
// ---------------------------------------------------------------------------
static __device__ __forceinline__ void tdm_load_tile_2d(
    unsigned lds_byte_off, const _Float16* gaddr,
    unsigned tensor_d0, unsigned tensor_d1,
    unsigned tile_d0, unsigned tile_d1, unsigned stride0) {
  unsigned long long ga = (unsigned long long)(uintptr_t)gaddr;
  u32x4 g0;
  g0[0] = rfl(1u);                                   // count=1, user mode
  g0[1] = rfl(lds_byte_off);                         // lds_addr
  g0[2] = rfl((unsigned)ga);                         // global_addr[31:0]
  g0[3] = rfl((unsigned)((ga >> 32) & 0x01FFFFFFull) // global_addr[56:32]
              | 0x80000000u);                        // type=2 (bits 127:126)
  i32x8 g1;
  g1[0] = (int)rfl(0x00010000u);                     // data_size=1 -> 2 bytes
  g1[1] = (int)rfl((tensor_d0 & 0xFFFFu) << 16);     // tensor_dim0 lo
  g1[2] = (int)rfl((tensor_d0 >> 16) | ((tensor_d1 & 0xFFFFu) << 16));
  g1[3] = (int)rfl((tensor_d1 >> 16) | (tile_d0 << 16));
  g1[4] = (int)rfl(tile_d1 & 0xFFFFu);               // tile_dim1 (tile_dim2=0)
  g1[5] = (int)rfl(stride0);                         // tensor_dim0_stride lo32
  g1[6] = 0;
  g1[7] = 0;
  i32x4 z4 = {0, 0, 0, 0};
#if defined(__clang_major__) && (__clang_major__ >= 23)
  i32x8 z8 = {0, 0, 0, 0, 0, 0, 0, 0};
  __builtin_amdgcn_tensor_load_to_lds(g0, g1, z4, z4, z8, 0);
#else
  __builtin_amdgcn_tensor_load_to_lds(g0, g1, z4, z4, 0);
#endif
}

// ---------------------------------------------------------------------------
// Kernel 0: one-time fp32 -> f16 conversion (bandwidth-trivial)
// ---------------------------------------------------------------------------
__global__ __launch_bounds__(256) void cvt_f16_kernel(
    const float* __restrict__ src, _Float16* __restrict__ dst, int n) {
  int i = (blockIdx.x * 256 + threadIdx.x) * 8;
  if (i >= n) return;
  f8 v = *(const f8*)(src + i);
  h8 o;
#pragma unroll
  for (int e = 0; e < 8; ++e) o[e] = (_Float16)v[e];
  *(h8*)(dst + i) = o;
}

// ---------------------------------------------------------------------------
// Kernel 1: fused Q/K/V projections, all-f16 operands.  C = X @ W^T + b
//   Q,K stored (b,h,t,dh); V stored transposed (b,h,dh,t). f16 outputs.
// ---------------------------------------------------------------------------
__global__ __launch_bounds__(256) void qkv_proj_kernel(
    const _Float16* __restrict__ xq, const _Float16* __restrict__ xk,
    const _Float16* __restrict__ xv,
    const _Float16* __restrict__ Wq, const _Float16* __restrict__ Wk,
    const _Float16* __restrict__ Wv,
    const float* __restrict__ bq, const float* __restrict__ bk,
    const float* __restrict__ bv,
    _Float16* __restrict__ Qf, _Float16* __restrict__ Kf,
    _Float16* __restrict__ Vtf) {
  const int lane = threadIdx.x & 31;
  const int wid  = threadIdx.x >> 5;
  int tile = blockIdx.x * 8 + wid;           // 512*32*3 tiles
  const int mt = tile & 511;  tile >>= 9;
  const int nt = tile & 31;
  const int which = tile >> 5;               // 0:Q 1:K 2:V

  const _Float16* X    = which == 0 ? xq : (which == 1 ? xk : xv);
  const _Float16* W    = which == 0 ? Wq : (which == 1 ? Wk : Wv);
  const float*    bias = which == 0 ? bq : (which == 1 ? bk : bv);

  const int m0 = mt << 4, n0 = nt << 4;
  const int li  = lane & 15;
  const int loa = (lane < 16) ? 0 : 16;
  const int lob = (lane < 16) ? 0 : 16;

  const _Float16* xb = X + (size_t)(m0 + li) * D_ + loa;
  const _Float16* wb = W + (size_t)(n0 + li) * D_ + lob;

  f8 acc = {};
#pragma unroll 8
  for (int k = 0; k < D_; k += 32)
    acc = wmma_f16(load_a_f16(xb + k), load_b_f16(wb + k), acc);

  const int ng = n0 + li;
  const float bvv = bias[ng];
  const int b = m0 >> 12;
  const int h = ng >> 6, d = ng & 63;
  const int loc = (lane < 16) ? 0 : 8;
  const int tbase = (m0 & (T_ - 1)) + loc;

  if (which < 2) {
    _Float16* out = (which == 0 ? Qf : Kf) +
                    (((size_t)(b * H_ + h)) * T_ + tbase) * DH_ + d;
#pragma unroll
    for (int r = 0; r < 8; ++r) out[(size_t)r * DH_] = (_Float16)(acc[r] + bvv);
  } else {
    _Float16* out = Vtf + (((size_t)(b * H_ + h)) * DH_ + d) * T_ + tbase;
    h8 o;
#pragma unroll
    for (int r = 0; r < 8; ++r) o[r] = (_Float16)(acc[r] + bvv);
    *(h8*)out = o;
  }
}

// ---------------------------------------------------------------------------
// Kernel 2: causal flash attention.
//   4 waves/WG cover 64 consecutive query rows of one (b,h). Wave 0 DMAs the
//   shared 32-key K/V^T block into a double-buffered 16KB LDS tile with the
//   Tensor Data Mover; one __syncthreads per block publishes it. Fragments
//   are read with ds_load via explicit addrspace(3) pointers.
//   S^T = K Q^T; O^T += V^T P^T; f32 online softmax.
// ---------------------------------------------------------------------------
__global__ __launch_bounds__(128) void attn_kernel(
    const _Float16* __restrict__ Q, const _Float16* __restrict__ K,
    const _Float16* __restrict__ Vt, _Float16* __restrict__ O) {
  __shared__ _Float16 smem[2][4096];   // [buf][K 32x64 | V^T 64x32] = 16 KB

  const int lane  = threadIdx.x & 31;
  const int wid   = threadIdx.x >> 5;
  const int tile0 = blockIdx.x * 4;          // first wave-tile of this WG
  const int bh    = tile0 >> 8;              // 256 q-tiles per (b,h)
  const int q0w   = (tile0 & 255) << 4;      // WG base query row
  const int q0    = q0w + wid * 16;          // this wave's 16 query rows

  const _Float16* Qb = Q  + (size_t)bh * T_ * DH_;
  const _Float16* Kb = K  + (size_t)bh * T_ * DH_;
  const _Float16* Vb = Vt + (size_t)bh * DH_ * T_;

  const int li  = lane & 15;
  const int loa = (lane < 16) ? 0 : 8;
  const int lob = (lane < 16) ? 0 : 16;
  const int ig  = q0 + li;                   // query row owned by this lane

  const _Float16* qp = Qb + (size_t)ig * DH_ + lob;
  const h16 bq0 = load_b_f16(qp);            // d 0..31
  const h16 bq1 = load_b_f16(qp + 32);       // d 32..63

  // LDS byte offsets of the two buffers (low 32 bits of generic address)
  const unsigned sm0 = (unsigned)(size_t)&smem[0][0];
  const unsigned sm1 = (unsigned)(size_t)&smem[1][0];

  f8 acc0 = {}, acc1 = {}, acc2 = {}, acc3 = {};
  float m = -1e30f, l = 0.0f;
  const int my_limit = q0 + 15;
  const int wg_limit = q0w + 63;             // last row handled by wave 3

  // wave 0 stages one 32-key block: K rows [jb,jb+32), V^T cols [jb,jb+32)
  auto stage = [&](unsigned sb, int jb) {
    tdm_load_tile_2d(sb,        Kb + (size_t)jb * DH_, DH_, T_, DH_, 32, DH_);
    tdm_load_tile_2d(sb + 4096, Vb + jb,               T_, DH_, 32, DH_, T_);
  };

  if (wid == 0) {
    stage(sm0, 0);
    __builtin_amdgcn_s_wait_tensorcnt(0);
  }
  __syncthreads();

  int buf = 0;
  for (int jb = 0; jb <= wg_limit; jb += 32) {
    const unsigned sb  = buf ? sm1 : sm0;
    const unsigned sbn = buf ? sm0 : sm1;
    const bool has_next = (jb + 32) <= wg_limit;
    if (wid == 0 && has_next) stage(sbn, jb + 32);   // overlap DMA w/ compute

    if (jb <= my_limit) {
      const unsigned k0 = sb + (unsigned)(li * DH_ + lob) * 2;         // j 0..15
      const unsigned k1 = sb + (unsigned)((16 + li) * DH_ + lob) * 2;  // j 16..31
      const unsigned vv = sb + 4096 + (unsigned)(li * 32 + lob) * 2;   // d-rows

      f8 s0 = {}, s1 = {};
      s0 = wmma_f16(load_a_lds(k0),      bq0, s0);
      s0 = wmma_f16(load_a_lds(k0 + 64), bq1, s0);   // +32 halves
      s1 = wmma_f16(load_a_lds(k1),      bq0, s1);
      s1 = wmma_f16(load_a_lds(k1 + 64), bq1, s1);

      // scale + causal mask + row max (row split between lanes li, li+16)
      float mx = -1e30f;
#pragma unroll
      for (int r = 0; r < 8; ++r) {
        const int j0 = jb + loa + r;
        const int j1 = j0 + 16;
        float v0 = (j0 <= ig) ? s0[r] * 0.125f : -1e30f;
        float v1 = (j1 <= ig) ? s1[r] * 0.125f : -1e30f;
        s0[r] = v0; s1[r] = v1;
        mx = fmaxf(mx, fmaxf(v0, v1));
      }
      mx = fmaxf(mx, __shfl_xor(mx, 16, 32));
      const float mn   = fmaxf(m, mx);
      const float corr = __expf(m - mn);
      m = mn;

      // exponentiate, row-sum, repack P^T into a B-fragment (partner swap)
      float ls = 0.0f;
      float own0[8], own1[8], other[8];
#pragma unroll
      for (int r = 0; r < 8; ++r) {
        float p0 = __expf(s0[r] - mn);
        float p1 = __expf(s1[r] - mn);
        own0[r] = p0; own1[r] = p1;
        ls += p0 + p1;
        other[r] = __shfl_xor((lane < 16) ? p1 : p0, 16, 32);
      }
      l = l * corr + ls + __shfl_xor(ls, 16, 32);

      h16 pt;
#pragma unroll
      for (int r = 0; r < 8; ++r) {
        if (lane < 16) { pt[r] = (_Float16)own0[r];  pt[r + 8] = (_Float16)other[r]; }
        else           { pt[r] = (_Float16)other[r]; pt[r + 8] = (_Float16)own1[r]; }
      }

#pragma unroll
      for (int r = 0; r < 8; ++r) {
        acc0[r] *= corr; acc1[r] *= corr; acc2[r] *= corr; acc3[r] *= corr;
      }
      acc0 = wmma_f16(load_a_lds(vv),        pt, acc0);  // d  0..15
      acc1 = wmma_f16(load_a_lds(vv + 1024), pt, acc1);  // d 16..31
      acc2 = wmma_f16(load_a_lds(vv + 2048), pt, acc2);  // d 32..47
      acc3 = wmma_f16(load_a_lds(vv + 3072), pt, acc3);  // d 48..63
    }

    if (wid == 0 && has_next) __builtin_amdgcn_s_wait_tensorcnt(0);
    __syncthreads();   // publish next buffer; everyone done reading current
    buf ^= 1;
  }

  const float inv = 1.0f / l;
  const int b = bh >> 3, h = bh & 7;
  const int loc = (lane < 16) ? 0 : 8;
  _Float16* ob = O + ((size_t)(b * T_ + ig)) * D_ + h * DH_ + loc;
  h8 o0, o1, o2, o3;
#pragma unroll
  for (int r = 0; r < 8; ++r) {
    o0[r] = (_Float16)(acc0[r] * inv);
    o1[r] = (_Float16)(acc1[r] * inv);
    o2[r] = (_Float16)(acc2[r] * inv);
    o3[r] = (_Float16)(acc3[r] * inv);
  }
  *(h8*)(ob)      = o0;
  *(h8*)(ob + 16) = o1;
  *(h8*)(ob + 32) = o2;
  *(h8*)(ob + 48) = o3;
}

// ---------------------------------------------------------------------------
// Kernel 3: output projection.  out(f32) = Of16 @ Wo^T + bo
// ---------------------------------------------------------------------------
__global__ __launch_bounds__(256) void out_proj_kernel(
    const _Float16* __restrict__ Of, const _Float16* __restrict__ Wo,
    const float* __restrict__ bo, float* __restrict__ out) {
  const int lane = threadIdx.x & 31;
  const int wid  = threadIdx.x >> 5;
  const int tile = blockIdx.x * 8 + wid;
  const int mt = tile & 511;
  const int nt = tile >> 9;
  const int m0 = mt << 4, n0 = nt << 4;
  const int li  = lane & 15;
  const int loa = (lane < 16) ? 0 : 16;
  const int lob = (lane < 16) ? 0 : 16;

  const _Float16* xb = Of + (size_t)(m0 + li) * D_ + loa;
  const _Float16* wb = Wo + (size_t)(n0 + li) * D_ + lob;

  f8 acc = {};
#pragma unroll 8
  for (int k = 0; k < D_; k += 32)
    acc = wmma_f16(load_a_f16(xb + k), load_b_f16(wb + k), acc);

  const int ng = n0 + li;
  const float bb = bo[ng];
  const int loc = (lane < 16) ? 0 : 8;
  float* ob = out + (size_t)(m0 + loc) * D_ + ng;
#pragma unroll
  for (int r = 0; r < 8; ++r) ob[(size_t)r * D_] = acc[r] + bb;
}

// ---------------------------------------------------------------------------
extern "C" void kernel_launch(void* const* d_in, const int* in_sizes, int n_in,
                              void* d_out, int out_size, void* d_ws,
                              size_t ws_size, hipStream_t stream) {
  const float* q_in = (const float*)d_in[0];
  const float* k_in = (const float*)d_in[1];
  const float* v_in = (const float*)d_in[2];
  // d_in[3] = mask (tril): causal mask applied analytically in attn_kernel
  const float* Wq = (const float*)d_in[4];
  const float* bq = (const float*)d_in[5];
  const float* Wk = (const float*)d_in[6];
  const float* bk = (const float*)d_in[7];
  const float* Wv = (const float*)d_in[8];
  const float* bv = (const float*)d_in[9];
  const float* Wo = (const float*)d_in[10];
  const float* bo = (const float*)d_in[11];

  _Float16* ws  = (_Float16*)d_ws;           // ~50 MB total
  _Float16* Qf  = ws;                        // projected Q (f16)
  _Float16* Kf  = Qf + NX_;
  _Float16* Vtf = Kf + NX_;
  _Float16* Xq  = Vtf + NX_;                 // f16 copies of inputs
  _Float16* Xk  = Xq + NX_;
  _Float16* Xv  = Xk + NX_;
  _Float16* Wqh = Xv + NX_;                  // f16 copies of weights
  _Float16* Wkh = Wqh + NW_;
  _Float16* Wvh = Wkh + NW_;
  _Float16* Woh = Wvh + NW_;
  _Float16* Of  = Xq;                        // alias: Xq is dead after proj

  const int gX = (int)(NX_ / (8 * 256));     // 2048 blocks
  const int gW = (int)(NW_ / (8 * 256));     // 128 blocks
  cvt_f16_kernel<<<gX, 256, 0, stream>>>(q_in, Xq, (int)NX_);
  cvt_f16_kernel<<<gX, 256, 0, stream>>>(k_in, Xk, (int)NX_);
  cvt_f16_kernel<<<gX, 256, 0, stream>>>(v_in, Xv, (int)NX_);
  cvt_f16_kernel<<<gW, 256, 0, stream>>>(Wq, Wqh, (int)NW_);
  cvt_f16_kernel<<<gW, 256, 0, stream>>>(Wk, Wkh, (int)NW_);
  cvt_f16_kernel<<<gW, 256, 0, stream>>>(Wv, Wvh, (int)NW_);
  cvt_f16_kernel<<<gW, 256, 0, stream>>>(Wo, Woh, (int)NW_);

  qkv_proj_kernel<<<6144, 256, 0, stream>>>(Xq, Xk, Xv, Wqh, Wkh, Wvh,
                                            bq, bk, bv, Qf, Kf, Vtf);
  attn_kernel<<<1024, 128, 0, stream>>>(Qf, Kf, Vtf, Of);
  out_proj_kernel<<<2048, 256, 0, stream>>>(Of, Woh, bo, (float*)d_out);
}
